// MultiheadAttention_50732153700451
// MI455X (gfx1250) — compile-verified, matrix operand order checked by reading, not run on hardware
//
#include <hip/hip_runtime.h>
#include <hip/hip_bf16.h>

// ---------------------------------------------------------------------------
// Multi-head attention forward for MI455X (gfx1250, wave32, WMMA bf16 path,
// TDM tensor_load_to_lds staging for the flash-attention K/V stream).
// B=2, S=2048, D=1024, H=16, DH=64.
// ---------------------------------------------------------------------------

#define BB 2
#define SS 2048
#define DD 1024
#define HH 16
#define DH 64
#define MTOK (BB * SS)   // 4096 token rows

typedef __attribute__((ext_vector_type(16))) __bf16 v16bf;
typedef __attribute__((ext_vector_type(8)))  __bf16 v8bf;
typedef __attribute__((ext_vector_type(8)))  float  v8f;
typedef __attribute__((ext_vector_type(4)))  unsigned int u32x4;
typedef __attribute__((ext_vector_type(8)))  int i32x8;
typedef __attribute__((ext_vector_type(4)))  int i32x4;

static __device__ __forceinline__ v8f wmma_bf16(v16bf a, v16bf b, v8f c) {
    // D = A(16x32) * B(32x16) + C, f32 accumulate
    return __builtin_amdgcn_wmma_f32_16x16x32_bf16(
        /*neg_a=*/false, a, /*neg_b=*/false, b,
        /*c_mod=*/(short)0, c, /*reuse_a=*/false, /*reuse_b=*/false);
}

static __device__ __forceinline__ v8bf ld8(const unsigned short* p) {
    return *reinterpret_cast<const v8bf*>(p);
}

static __device__ __forceinline__ v16bf cat16(v8bf lo, v8bf hi) {
    union { v16bf v; v8bf h[2]; } u;
    u.h[0] = lo; u.h[1] = hi;
    return u.v;
}

static __device__ __forceinline__ unsigned short f2bf(float f) {
    unsigned u = __builtin_bit_cast(unsigned, f);
    u += 0x7FFFu + ((u >> 16) & 1u);          // round-to-nearest-even
    return (unsigned short)(u >> 16);
}

// ---------------------------------------------------------------------------
// Tensor Data Mover: 2D tile load global -> LDS (bf16 elements).
// D# group0: count=1, lds_addr, 57-bit global_addr, type=2 (image).
// D# group1: data_size=1 (2 bytes), tensor dims/strides, tile dims.
// Groups 2/3 disabled (2D tensor); 6-arg builtin (clang-23 flavor).
// ---------------------------------------------------------------------------
static __device__ __forceinline__ void tdm_load_2d(
    unsigned lds_off, const void* gaddr,
    unsigned tensor_d0, unsigned tensor_d1,       // tensor extents (elements)
    unsigned tile_d0, unsigned tile_d1,           // tile extents (elements)
    unsigned long long stride0)                   // dim0 stride (elements)
{
    const unsigned long long ga = (unsigned long long)gaddr;
    u32x4 g0;
    g0[0] = 0x1u;                                     // count=1, user mode
    g0[1] = lds_off;                                  // LDS byte address
    g0[2] = (unsigned)(ga & 0xFFFFFFFFu);             // global_addr[31:0]
    g0[3] = (unsigned)((ga >> 32) & 0x01FFFFFFu)      // global_addr[56:32]
          | (2u << 30);                               // type=2
    i32x8 g1;
    g1[0] = (int)(1u << 16);                          // data_size=2B, wg_mask=0
    g1[1] = (int)((tensor_d0 & 0xFFFFu) << 16);       // tensor_dim0[15:0]
    g1[2] = (int)(((tensor_d0 >> 16) & 0xFFFFu)
          | ((tensor_d1 & 0xFFFFu) << 16));           // dim0 hi / dim1 lo
    g1[3] = (int)(((tensor_d1 >> 16) & 0xFFFFu)
          | ((tile_d0 & 0xFFFFu) << 16));             // dim1 hi / tile_dim0
    g1[4] = (int)(tile_d1 & 0xFFFFu);                 // tile_dim1, tile_dim2=0
    g1[5] = (int)(stride0 & 0xFFFFFFFFu);             // dim0_stride[31:0]
    g1[6] = (int)((stride0 >> 32) & 0xFFFFu);         // dim0_stride[47:32]
    g1[7] = 0;
    const i32x4 z4 = {0, 0, 0, 0};
    const i32x8 z8 = {0, 0, 0, 0, 0, 0, 0, 0};
    __builtin_amdgcn_tensor_load_to_lds(g0, g1, z4, z4, z8, 0);
}

// ---------------------------------------------------------------------------
// Stage 0: fp32 -> bf16 conversion (activations + weights)
// ---------------------------------------------------------------------------
__global__ void cvt_f32_bf16(const float* __restrict__ src,
                             unsigned short* __restrict__ dst, int n) {
    int i = (blockIdx.x * blockDim.x + threadIdx.x) * 4;
    if (i + 3 < n) {
        float4 v = *reinterpret_cast<const float4*>(src + i);
        dst[i + 0] = f2bf(v.x);
        dst[i + 1] = f2bf(v.y);
        dst[i + 2] = f2bf(v.z);
        dst[i + 3] = f2bf(v.w);
    }
}

// ---------------------------------------------------------------------------
// Stage 1/2/5: tiled GEMM  y[m][n] = sum_k X[m][k] * W[n][k]  (+ bias)
// One wave computes a 16x64 output tile. K streamed in steps of 32.
// MODE 0: store bf16 [B][H][S][DH]; MODE 1: bf16 [B][H][DH][S] (V^T);
// MODE 2: fp32 [m][n] + bias (final output projection)
// ---------------------------------------------------------------------------
template <int MODE>
__global__ __launch_bounds__(256) void proj_gemm(
    const unsigned short* __restrict__ X,   // [4096][1024] bf16
    const unsigned short* __restrict__ W,   // [1024][1024] bf16 (row n, col k)
    const float* __restrict__ bias,         // [1024]
    void* __restrict__ outp)
{
    const int lane = threadIdx.x & 31;
    const int wave = threadIdx.x >> 5;
    const int task = blockIdx.x * 8 + wave;       // 4096 tasks
    const int mt   = task >> 4;                   // 0..255  (16-row tile)
    const int ng   = task & 15;                   // 0..15   (64-col group)
    const int hlf  = lane >> 4;
    const int l16  = lane & 15;
    const int koff = hlf * 8;

    const unsigned short* xrow = X + (mt * 16 + l16) * DD;
    const int n0 = ng * 64 + l16;
    const unsigned short* w0 = W + (n0 +  0) * DD;
    const unsigned short* w1 = W + (n0 + 16) * DD;
    const unsigned short* w2 = W + (n0 + 32) * DD;
    const unsigned short* w3 = W + (n0 + 48) * DD;

    v8f acc0 = {}, acc1 = {}, acc2 = {}, acc3 = {};

    for (int k0 = 0; k0 < DD; k0 += 32) {
        __builtin_prefetch(w0 + k0 + 64, 0, 1);   // global_prefetch_b8
        __builtin_prefetch(xrow + k0 + 64, 0, 1);
        v16bf a  = cat16(ld8(xrow + k0 + koff), ld8(xrow + k0 + koff + 16));
        v16bf b0 = cat16(ld8(w0 + k0 + koff),   ld8(w0 + k0 + koff + 16));
        v16bf b1 = cat16(ld8(w1 + k0 + koff),   ld8(w1 + k0 + koff + 16));
        v16bf b2 = cat16(ld8(w2 + k0 + koff),   ld8(w2 + k0 + koff + 16));
        v16bf b3 = cat16(ld8(w3 + k0 + koff),   ld8(w3 + k0 + koff + 16));
        acc0 = wmma_bf16(a, b0, acc0);
        acc1 = wmma_bf16(a, b1, acc1);
        acc2 = wmma_bf16(a, b2, acc2);
        acc3 = wmma_bf16(a, b3, acc3);
    }

    // C/D layout: VGPR r holds row mt*16 + r + hlf*8, column = tile_n + l16
    const float bn0 = bias[ng * 64 +  0 + l16];
    const float bn1 = bias[ng * 64 + 16 + l16];
    const float bn2 = bias[ng * 64 + 32 + l16];
    const float bn3 = bias[ng * 64 + 48 + l16];

#pragma unroll
    for (int r = 0; r < 8; ++r) {
        const int m = mt * 16 + r + hlf * 8;      // global token row
        const int b = m >> 11;                    // /2048
        const int s = m & (SS - 1);
        const float v0 = acc0[r] + bn0;
        const float v1 = acc1[r] + bn1;
        const float v2 = acc2[r] + bn2;
        const float v3 = acc3[r] + bn3;
        if constexpr (MODE == 2) {
            float* out = (float*)outp;
            float* orow = out + m * DD + ng * 64 + l16;
            orow[0] = v0; orow[16] = v1; orow[32] = v2; orow[48] = v3;
        } else {
            unsigned short* out = (unsigned short*)outp;
#pragma unroll
            for (int nt = 0; nt < 4; ++nt) {
                const int n  = ng * 64 + nt * 16 + l16;
                const int h  = n >> 6;
                const int dh = n & (DH - 1);
                const float v = (nt == 0) ? v0 : (nt == 1) ? v1 : (nt == 2) ? v2 : v3;
                long idx;
                if constexpr (MODE == 0)
                    idx = (((long)(b * HH + h) * SS + s) * DH + dh);
                else
                    idx = (((long)(b * HH + h) * DH + dh) * SS + s);
                out[idx] = f2bf(v);
            }
        }
    }
}

// ---------------------------------------------------------------------------
// Stage 3: flash attention. One wave owns a 16-query tile of one (b,h).
// K/V tiles are staged into LDS by the Tensor Data Mover (double-buffered,
// TENSORcnt-synchronized), WMMA operands are fed from LDS (ds_load_b128).
// ---------------------------------------------------------------------------
#define WPB 4   // waves per block

__global__ __launch_bounds__(128) void attn_flash(
    const unsigned short* __restrict__ Qh,  // [B*H][S][64]
    const unsigned short* __restrict__ Kh,  // [B*H][S][64]
    const unsigned short* __restrict__ Vt,  // [B*H][64][S]
    unsigned short* __restrict__ AO)        // [B*S][1024] bf16
{
    __shared__ __align__(16) unsigned short kbuf[WPB][2][32 * DH]; // 32 keys x 64 dh
    __shared__ __align__(16) unsigned short vbuf[WPB][2][DH * 32]; // 64 dh x 32 keys
    __shared__ __align__(16) unsigned short pbuf[WPB][16 * 32];

    const int lane = threadIdx.x & 31;
    const int wave = threadIdx.x >> 5;
    const int task = blockIdx.x * WPB + wave; // B*H*(S/16) = 4096
    const int qt   = task & 127;              // query tile
    const int bh   = task >> 7;               // 0..31
    const int hlf  = lane >> 4;
    const int l16  = lane & 15;
    const int koff = hlf * 8;
    const float sc = 0.125f;                  // 1/sqrt(64)

    const unsigned short* kbh = Kh + (long)bh * SS * DH;
    const unsigned short* vbh = Vt + (long)bh * DH * SS;

    // Q tile as A operand (held in registers for whole kernel)
    const unsigned short* qbase = Qh + ((long)bh * SS + qt * 16 + l16) * DH;
    const v16bf qlo = cat16(ld8(qbase + koff),      ld8(qbase + koff + 16));
    const v16bf qhi = cat16(ld8(qbase + 32 + koff), ld8(qbase + 32 + koff + 16));

    float mrun[8], lrun[8];
#pragma unroll
    for (int r = 0; r < 8; ++r) { mrun[r] = -1e30f; lrun[r] = 0.0f; }
    v8f acc[4] = {};

    unsigned short* P = pbuf[wave];

    // TDM issue for one 32-key tile: K rows [kt..kt+32) and V^T cols [kt..kt+32)
    auto issue_tile = [&](int kt, int bufIdx) {
        tdm_load_2d((unsigned)(unsigned long long)(const void*)&kbuf[wave][bufIdx][0],
                    kbh + (long)kt * DH,
                    /*tensor_d0=*/DH, /*tensor_d1=*/SS - kt,
                    /*tile_d0=*/DH,   /*tile_d1=*/32,
                    /*stride0=*/DH);
        tdm_load_2d((unsigned)(unsigned long long)(const void*)&vbuf[wave][bufIdx][0],
                    vbh + kt,
                    /*tensor_d0=*/SS - kt, /*tensor_d1=*/DH,
                    /*tile_d0=*/32,        /*tile_d1=*/DH,
                    /*stride0=*/SS);
    };

    issue_tile(0, 0);
    int cur = 0;

    for (int kt = 0; kt < SS; kt += 32) {
        // Prefetch next tile via TDM, then wait for the current one
        // (TDM ops from one wave complete in order; <=2 outstanding ==
        //  the 2 descriptors of the current tile are done).
        if (kt + 32 < SS) {
            issue_tile(kt + 32, cur ^ 1);
            __builtin_amdgcn_s_wait_tensorcnt(2);
        } else {
            __builtin_amdgcn_s_wait_tensorcnt(0);
        }
        asm volatile("" ::: "memory");   // keep LDS reads below the wait

        const unsigned short* kt_lds = &kbuf[wave][cur][0]; // [32][64]
        const unsigned short* vt_lds = &vbuf[wave][cur][0]; // [64][32]

        // ---- scores: S = Q * K^T, two 16-key column tiles ----
        const unsigned short* k0p = kt_lds + l16 * DH;
        const unsigned short* k1p = k0p + 16 * DH;
        v16bf b0lo = cat16(ld8(k0p + koff),      ld8(k0p + koff + 16));
        v16bf b0hi = cat16(ld8(k0p + 32 + koff), ld8(k0p + 32 + koff + 16));
        v16bf b1lo = cat16(ld8(k1p + koff),      ld8(k1p + koff + 16));
        v16bf b1hi = cat16(ld8(k1p + 32 + koff), ld8(k1p + 32 + koff + 16));
        v8f z = {};
        v8f s0 = wmma_bf16(qhi, b0hi, z); s0 = wmma_bf16(qlo, b0lo, s0);
        v8f s1 = wmma_bf16(qhi, b1hi, z); s1 = wmma_bf16(qlo, b1lo, s1);

        // ---- online softmax over 32 new columns ----
        float alpha[8];
#pragma unroll
        for (int r = 0; r < 8; ++r) {
            float e0 = s0[r] * sc, e1 = s1[r] * sc;
            float mx = fmaxf(e0, e1);
#pragma unroll
            for (int msk = 8; msk >= 1; msk >>= 1)   // reduce within 16-lane half
                mx = fmaxf(mx, __shfl_xor(mx, msk, 32));
            const float mn = fmaxf(mrun[r], mx);
            const float al = __expf(mrun[r] - mn);
            const float p0 = __expf(e0 - mn);
            const float p1 = __expf(e1 - mn);
            float rs = p0 + p1;
#pragma unroll
            for (int msk = 8; msk >= 1; msk >>= 1)
                rs += __shfl_xor(rs, msk, 32);
            lrun[r] = lrun[r] * al + rs;
            mrun[r] = mn;
            alpha[r] = al;
            const int row = r + hlf * 8;             // C/D row for this reg
            P[row * 32 + l16]      = f2bf(p0);
            P[row * 32 + 16 + l16] = f2bf(p1);
        }
#pragma unroll
        for (int nt = 0; nt < 4; ++nt)
#pragma unroll
            for (int r = 0; r < 8; ++r) acc[nt][r] *= alpha[r];

        __syncthreads();   // order ds_store (C/D layout) before ds_load (A layout)

        // ---- P as A operand (16x32), V^T tiles from LDS as B operand ----
        const unsigned short* prow = P + l16 * 32 + koff;
        v16bf pa = cat16(ld8(prow), ld8(prow + 16));
#pragma unroll
        for (int nt = 0; nt < 4; ++nt) {
            const unsigned short* vr = vt_lds + (nt * 16 + l16) * 32 + koff;
            v16bf vb = cat16(ld8(vr), ld8(vr + 16));
            acc[nt] = wmma_bf16(pa, vb, acc[nt]);
        }
        cur ^= 1;
    }

    // ---- normalize and store attention output [B][S][H*DH] bf16 ----
    const int b = bh >> 4, h = bh & 15;
#pragma unroll
    for (int r = 0; r < 8; ++r) {
        const float inv = 1.0f / lrun[r];
        const int s = qt * 16 + r + hlf * 8;
        unsigned short* orow = AO + ((long)(b * SS + s)) * DD + h * DH;
        orow[ 0 + l16] = f2bf(acc[0][r] * inv);
        orow[16 + l16] = f2bf(acc[1][r] * inv);
        orow[32 + l16] = f2bf(acc[2][r] * inv);
        orow[48 + l16] = f2bf(acc[3][r] * inv);
    }
}

// ---------------------------------------------------------------------------
// Host-side launch sequence (graph-capture safe: only kernel launches)
// ---------------------------------------------------------------------------
extern "C" void kernel_launch(void* const* d_in, const int* in_sizes, int n_in,
                              void* d_out, int out_size, void* d_ws, size_t ws_size,
                              hipStream_t stream) {
    const float* x_q  = (const float*)d_in[0];
    const float* x_kv = (const float*)d_in[1];
    const float* Wq   = (const float*)d_in[2];
    const float* bq   = (const float*)d_in[3];
    const float* Wk   = (const float*)d_in[4];
    const float* bk   = (const float*)d_in[5];
    const float* Wv   = (const float*)d_in[6];
    const float* bv   = (const float*)d_in[7];
    const float* Wo   = (const float*)d_in[8];
    const float* bo   = (const float*)d_in[9];
    float* out = (float*)d_out;

    char* ws = (char*)d_ws;
    const size_t MB = 1u << 20;
    unsigned short* Xq  = (unsigned short*)(ws +  0 * MB);  // 8 MB
    unsigned short* Xkv = (unsigned short*)(ws +  8 * MB);  // 8 MB
    unsigned short* Wqb = (unsigned short*)(ws + 16 * MB);  // 2 MB
    unsigned short* Wkb = (unsigned short*)(ws + 18 * MB);
    unsigned short* Wvb = (unsigned short*)(ws + 20 * MB);
    unsigned short* Wob = (unsigned short*)(ws + 22 * MB);
    unsigned short* Qh  = (unsigned short*)(ws + 24 * MB);  // 8 MB
    unsigned short* Kh  = (unsigned short*)(ws + 32 * MB);  // 8 MB
    unsigned short* Vt  = (unsigned short*)(ws + 40 * MB);  // 8 MB (transposed)
    unsigned short* AO  = (unsigned short*)(ws + 48 * MB);  // 8 MB

    const int nX = MTOK * DD;        // 4M elements
    const int nW = DD * DD;          // 1M elements
    cvt_f32_bf16<<<nX / 1024, 256, 0, stream>>>(x_q,  Xq,  nX);
    cvt_f32_bf16<<<nX / 1024, 256, 0, stream>>>(x_kv, Xkv, nX);
    cvt_f32_bf16<<<nW / 1024, 256, 0, stream>>>(Wq, Wqb, nW);
    cvt_f32_bf16<<<nW / 1024, 256, 0, stream>>>(Wk, Wkb, nW);
    cvt_f32_bf16<<<nW / 1024, 256, 0, stream>>>(Wv, Wvb, nW);
    cvt_f32_bf16<<<nW / 1024, 256, 0, stream>>>(Wo, Wob, nW);

    // 4096 wave-tasks / 8 waves per block = 512 blocks
    proj_gemm<0><<<512, 256, 0, stream>>>(Xq,  Wqb, bq, (void*)Qh);
    proj_gemm<0><<<512, 256, 0, stream>>>(Xkv, Wkb, bk, (void*)Kh);
    proj_gemm<1><<<512, 256, 0, stream>>>(Xkv, Wvb, bv, (void*)Vt);

    // 4096 wave-tasks / 4 waves per block = 1024 blocks
    attn_flash<<<1024, 128, 0, stream>>>(Qh, Kh, Vt, AO);

    proj_gemm<2><<<512, 256, 0, stream>>>(AO, Wob, bo, (void*)out);
}
